// MultiGRU_66451734003826
// MI455X (gfx1250) — matrix operand order repaired and checked by reading, not compiled, not run on hardware
//
#include <hip/hip_runtime.h>
#include <hip/hip_bf16.h>
#include <math.h>

typedef __bf16 bf16;
typedef __attribute__((ext_vector_type(16))) __bf16 v16bf;
typedef __attribute__((ext_vector_type(8)))  __bf16 v8bf;
typedef __attribute__((ext_vector_type(8)))  float  v8f;

#define HID 256
#define FIN 11
#define MT  64          // rows per block
#define SB  264         // LDS row stride (bf16 elems); 528B, not a multiple of 256B

__device__ __forceinline__ float sigm(float x) { return 1.0f / (1.0f + __expf(-x)); }

// A fragment (16x32 bf16, M x K) from a row-major LDS tile with stride SB.
// Contiguous K order (matches load_b): lanes 0-15 hold K=[0..15], lanes 16-31 hold K=[16..31].
__device__ __forceinline__ v16bf load_a(const bf16* tile, int mt, int kk, int lane) {
    int half = lane >> 4;
    int row  = mt * 16 + (lane & 15);
    const bf16* p = tile + row * SB + kk * 32 + 16 * half;
    v8bf lo = *(const v8bf*)(p);
    v8bf hi = *(const v8bf*)(p + 8);
    return __builtin_shufflevector(lo, hi, 0,1,2,3,4,5,6,7,8,9,10,11,12,13,14,15);
}

// B fragment (32x16 bf16, K x N), B = W^T with W row-major [256 out][256 in] bf16.
// Lane covers col n = lane%16, K = kk*32 + 16*(lane/16) .. +15  -> 32B contiguous.
__device__ __forceinline__ v16bf load_b(const bf16* __restrict__ W, int nt, int kk, int lane) {
    int half = lane >> 4;
    int col  = nt * 16 + (lane & 15);
    return *(const v16bf*)(W + (size_t)col * HID + kk * 32 + 16 * half);
}

__device__ __forceinline__ v8f wmma_bf16(v16bf a, v16bf b, v8f c) {
    return __builtin_amdgcn_wmma_f32_16x16x32_bf16(false, a, false, b, (short)0, c,
                                                   false, false);
}

__global__ void convert_w_kernel(const float* __restrict__ a,   // bb_Wh  [3*256*256]
                                 const float* __restrict__ b,   // u_Wx   [3*256*256]
                                 const float* __restrict__ c,   // sp_Wx  [3*256*256]
                                 bf16* __restrict__ out) {      // [9*65536]
    int i = blockIdx.x * 256 + threadIdx.x;
    float v;
    if (i < 196608)       v = a[i];
    else if (i < 393216)  v = b[i - 196608];
    else                  v = c[i - 393216];
    out[i] = (bf16)v;
}

__global__ void zero_h_kernel(bf16* __restrict__ H) {
    int i = blockIdx.x * 256 + threadIdx.x;   // chunk of 8 bf16
    v8bf z = {};
    *(v8bf*)(H + (size_t)i * 8) = z;
}

__global__ __launch_bounds__(256, 1) void gru_step_kernel(
    const float* __restrict__ Xt,        // [N,11] for this timestep
    bf16* __restrict__ Hstate,           // [N,256] in/out (in-place per row-tile)
    const bf16* __restrict__ Wbf,        // 9 x [256,256] bf16: bbWh0..2, uWx0..2, spWx0..2
    const float* __restrict__ bbWx,      // [3,256,11]
    const float* __restrict__ bb_bx, const float* __restrict__ bb_bh,
    const float* __restrict__ u_bx,  const float* __restrict__ u_bh,
    const float* __restrict__ sp_bx, const float* __restrict__ sp_bh,
    const float* __restrict__ Wu, const float* __restrict__ bu,
    const float* __restrict__ Ws, const float* __restrict__ bs,
    const float* __restrict__ Wp, const float* __restrict__ bp,
    float* __restrict__ outU, float* __restrict__ outS, float* __restrict__ outP,
    int n)
{
    __shared__ bf16 BH[MT * SB];   // Hprev, later hu
    __shared__ bf16 BZ[MT * SB];   // z,     later hp
    __shared__ bf16 BR[MT * SB];   // r * Hprev
    __shared__ bf16 BN[MT * SB];   // Hn
    __shared__ float XT[MT * 12];  // X tile, stride 12

    const int tid  = threadIdx.x;
    const int lane = tid & 31;
    const int wave = tid >> 5;
    const int m0   = blockIdx.x * MT;
    const int half = lane >> 4;
    const int ncol = lane & 15;

    // ---- stage Hprev (bf16) and X tile (X is read-once: non-temporal) ----
    for (int i = tid; i < MT * (HID / 8); i += 256) {
        int row = i >> 5;
        int c   = (i & 31) * 8;
        int gm  = m0 + row; if (gm > n - 1) gm = n - 1;
        *(v8bf*)(BH + row * SB + c) = *(const v8bf*)(Hstate + (size_t)gm * HID + c);
    }
    for (int i = tid; i < MT * FIN; i += 256) {
        int row = i / FIN, k = i - row * FIN;
        int gm = m0 + row; if (gm > n - 1) gm = n - 1;
        XT[row * 12 + k] = __builtin_nontemporal_load(Xt + (size_t)gm * FIN + k);
    }
    __syncthreads();

    // ---- phase 1: z & r gates, two independent WMMA chains sharing the A fragment ----
    for (int j = wave; j < 64; j += 8) {
        int mt = j >> 4;
        int nt = j & 15;
        const bf16* W0 = Wbf;                       // bb_Wh[0]
        const bf16* W1 = Wbf + (size_t)65536;       // bb_Wh[1]
        v8f acc0 = {}, acc1 = {};
#pragma unroll
        for (int kk = 0; kk < 8; ++kk) {
            v16bf a  = load_a(BH, mt, kk, lane);
            v16bf b0 = load_b(W0, nt, kk, lane);
            acc0 = wmma_bf16(a, b0, acc0);
            v16bf b1 = load_b(W1, nt, kk, lane);
            acc1 = wmma_bf16(a, b1, acc1);
        }
        int nn = nt * 16 + ncol;
        float w0[FIN], w1[FIN];
#pragma unroll
        for (int k = 0; k < FIN; ++k) w0[k] = bbWx[(0 * HID + nn) * FIN + k];
#pragma unroll
        for (int k = 0; k < FIN; ++k) w1[k] = bbWx[(1 * HID + nn) * FIN + k];
        float bias0 = bb_bx[nn] + bb_bh[nn];
        float bias1 = bb_bx[HID + nn] + bb_bh[HID + nn];
#pragma unroll
        for (int r = 0; r < 8; ++r) {
            int m = mt * 16 + r + 8 * half;
            float x0 = bias0, x1 = bias1;
#pragma unroll
            for (int k = 0; k < FIN; ++k) {
                float xv = XT[m * 12 + k];
                x0 += xv * w0[k];
                x1 += xv * w1[k];
            }
            float z  = sigm(acc0[r] + x0);
            float rr = sigm(acc1[r] + x1);
            BZ[m * SB + nn] = (bf16)z;
            BR[m * SB + nn] = (bf16)(rr * (float)BH[m * SB + nn]);
        }
    }
    __syncthreads();

    // ---- phase 2: h~ GEMM, two ntiles per job sharing the A fragment ----
    for (int j = wave; j < 32; j += 8) {
        int mt  = j >> 3;
        int nt0 = (j & 7) * 2, nt1 = nt0 + 1;
        const bf16* W2 = Wbf + (size_t)2 * 65536;   // bb_Wh[2]
        v8f accA = {}, accB = {};
#pragma unroll
        for (int kk = 0; kk < 8; ++kk) {
            v16bf a  = load_a(BR, mt, kk, lane);
            v16bf bA = load_b(W2, nt0, kk, lane);
            accA = wmma_bf16(a, bA, accA);
            v16bf bB = load_b(W2, nt1, kk, lane);
            accB = wmma_bf16(a, bB, accB);
        }
#pragma unroll
        for (int s = 0; s < 2; ++s) {
            int nn = (s ? nt1 : nt0) * 16 + ncol;
            const v8f& acc = s ? accB : accA;
            float w[FIN];
#pragma unroll
            for (int k = 0; k < FIN; ++k) w[k] = bbWx[(2 * HID + nn) * FIN + k];
            float bias = bb_bx[2 * HID + nn] + bb_bh[2 * HID + nn];
#pragma unroll
            for (int r = 0; r < 8; ++r) {
                int m = mt * 16 + r + 8 * half;
                float xg = bias;
#pragma unroll
                for (int k = 0; k < FIN; ++k) xg += XT[m * 12 + k] * w[k];
                float ht = tanhf(acc[r] + xg);
                float z  = (float)BZ[m * SB + nn];
                float hn = z * (float)BH[m * SB + nn] + (1.0f - z) * ht;
                BN[m * SB + nn] = (bf16)hn;
            }
        }
    }
    __syncthreads();

    // ---- phase 3: hu & hp, four independent WMMA chains sharing the A fragment ----
    // (gru_u/gru_sp get H=0 in the reference: Wh terms vanish, r gate is dead)
    for (int j = wave; j < 64; j += 8) {
        int mt = j >> 4, nt = j & 15;
        const bf16* Wu0 = Wbf + (size_t)3 * 65536;  // u_Wx[0]
        const bf16* Wu2 = Wbf + (size_t)5 * 65536;  // u_Wx[2]
        const bf16* Ws0 = Wbf + (size_t)6 * 65536;  // sp_Wx[0]
        const bf16* Ws2 = Wbf + (size_t)8 * 65536;  // sp_Wx[2]
        v8f aU0 = {}, aU2 = {}, aS0 = {}, aS2 = {};
#pragma unroll
        for (int kk = 0; kk < 8; ++kk) {
            v16bf a = load_a(BN, mt, kk, lane);
            v16bf b;
            b = load_b(Wu0, nt, kk, lane); aU0 = wmma_bf16(a, b, aU0);
            b = load_b(Wu2, nt, kk, lane); aU2 = wmma_bf16(a, b, aU2);
            b = load_b(Ws0, nt, kk, lane); aS0 = wmma_bf16(a, b, aS0);
            b = load_b(Ws2, nt, kk, lane); aS2 = wmma_bf16(a, b, aS2);
        }
        int nn = nt * 16 + ncol;
        float bu0 = u_bx[nn] + u_bh[nn];
        float bu2 = u_bx[2 * HID + nn] + u_bh[2 * HID + nn];
        float bp0 = sp_bx[nn] + sp_bh[nn];
        float bp2 = sp_bx[2 * HID + nn] + sp_bh[2 * HID + nn];
#pragma unroll
        for (int r = 0; r < 8; ++r) {
            int m = mt * 16 + r + 8 * half;
            float hu = (1.0f - sigm(aU0[r] + bu0)) * tanhf(aU2[r] + bu2);
            float hp = (1.0f - sigm(aS0[r] + bp0)) * tanhf(aS2[r] + bp2);
            BH[m * SB + nn] = (bf16)hu;
            BZ[m * SB + nn] = (bf16)hp;
        }
    }
    __syncthreads();

    // ---- Hn writeback (re-read next step: keep regular-temporal, L2-resident) ----
    for (int i = tid; i < MT * (HID / 8); i += 256) {
        int row = i >> 5;
        int c   = (i & 31) * 8;
        int gm  = m0 + row;
        if (gm < n)
            *(v8bf*)(Hstate + (size_t)gm * HID + c) = *(const v8bf*)(BN + row * SB + c);
    }

    // ---- heads: u = hu@Wu^T+bu (3), s = hp.Ws+bs, p = hp.Wp+bp (write-once: NT stores) ----
    for (int task = tid; task < MT * 5; task += 256) {
        int row = task / 5, h = task - row * 5;
        int gm = m0 + row;
        if (gm >= n) continue;
        const bf16*  src = (h < 3) ? (BH + row * SB) : (BZ + row * SB);
        const float* wv  = (h == 0) ? Wu : (h == 1) ? Wu + HID : (h == 2) ? Wu + 2 * HID
                         : (h == 3) ? Ws : Wp;
        float acc = 0.0f;
        for (int k = 0; k < HID; ++k) acc += (float)src[k] * wv[k];
        if (h < 3)       __builtin_nontemporal_store(acc + bu[h], outU + (size_t)gm * 3 + h);
        else if (h == 3) __builtin_nontemporal_store(acc + bs[0], outS + gm);
        else             __builtin_nontemporal_store(acc + bp[0], outP + gm);
    }
}

extern "C" void kernel_launch(void* const* d_in, const int* in_sizes, int n_in,
                              void* d_out, int out_size, void* d_ws, size_t ws_size,
                              hipStream_t stream) {
    const float* X_seq = (const float*)d_in[0];
    // d_in[1] = edge (unused: ChebConv K=1 ignores edge_index)
    const float* bbWx  = (const float*)d_in[2];
    const float* bb_bx = (const float*)d_in[3];
    const float* bbWh  = (const float*)d_in[4];
    const float* bb_bh = (const float*)d_in[5];
    const float* uWx   = (const float*)d_in[6];
    const float* u_bx  = (const float*)d_in[7];
    // d_in[8] = u_Wh (dead: multiplied by zero state)
    const float* u_bh  = (const float*)d_in[9];
    const float* spWx  = (const float*)d_in[10];
    const float* sp_bx = (const float*)d_in[11];
    // d_in[12] = sp_Wh (dead)
    const float* sp_bh = (const float*)d_in[13];
    const float* Wu = (const float*)d_in[14];
    const float* bu = (const float*)d_in[15];
    const float* Ws = (const float*)d_in[16];
    const float* bs = (const float*)d_in[17];
    const float* Wp = (const float*)d_in[18];
    const float* bp = (const float*)d_in[19];
    float* out = (float*)d_out;

    const int T = 48, N = 50000;

    bf16* Hstate = (bf16*)d_ws;                                            // 25.6 MB
    bf16* Wbf    = (bf16*)((char*)d_ws + (size_t)N * HID * sizeof(bf16));  // +1.18 MB

    convert_w_kernel<<<2304, 256, 0, stream>>>(bbWh, uWx, spWx, Wbf);
    zero_h_kernel<<<(N * HID / 8 + 255) / 256, 256, 0, stream>>>(Hstate);

    const int nblk = (N + MT - 1) / MT;   // 782
    for (int t = 0; t < T; ++t) {
        gru_step_kernel<<<nblk, 256, 0, stream>>>(
            X_seq + (size_t)t * N * FIN, Hstate, Wbf,
            bbWx, bb_bx, bb_bh, u_bx, u_bh, sp_bx, sp_bh,
            Wu, bu, Ws, bs, Wp, bp,
            out + (size_t)t * N * 3,
            out + (size_t)T * N * 3 + (size_t)t * N,
            out + (size_t)T * N * 4 + (size_t)t * N,
            N);
    }
}
